// RobustAttention_23210003267797
// MI455X (gfx1250) — matrix operand order repaired and checked by reading, not compiled, not run on hardware
//
#include <hip/hip_runtime.h>

// ---- problem constants ----
#define BB 2
#define NN 2048
#define CC 1024
#define HH 16
#define DD 64
#define MM (BB * NN)   // 4096 rows

typedef __attribute__((ext_vector_type(8)))  __bf16 v8bf;
typedef __attribute__((ext_vector_type(16))) __bf16 v16bf;
typedef __attribute__((ext_vector_type(8)))  float  v8f;

union ABFrag { v16bf v; v8bf h[2]; };

__device__ __forceinline__ unsigned short f32_to_bf16_rne(float f) {
  union { float f; unsigned u; } x; x.f = f;
  unsigned r = x.u + 0x7FFFu + ((x.u >> 16) & 1u);
  return (unsigned short)(r >> 16);
}

__device__ __forceinline__ v8bf ld8(const unsigned short* p) {
  return *(const v8bf*)p;
}

__device__ __forceinline__ v8f wmma_bf16(const ABFrag& a, const ABFrag& b, v8f c) {
  // D = A(16x32 bf16) x B(32x16 bf16) + C(16x16 f32)
  return __builtin_amdgcn_wmma_f32_16x16x32_bf16(
      /*neg_a=*/false, a.v, /*neg_b=*/false, b.v,
      /*c_mod=*/(short)0, c, /*reuse_a=*/false, /*reuse_b=*/false);
}

// ---------------------------------------------------------------------------
// f32 -> bf16 conversion, 4 elements/thread
// ---------------------------------------------------------------------------
__global__ void cvt_f32_bf16(const float* __restrict__ src,
                             unsigned short* __restrict__ dst, int n4) {
  int i = blockIdx.x * blockDim.x + threadIdx.x;
  if (i >= n4) return;
  float4 f = ((const float4*)src)[i];
  ushort4 o;
  o.x = f32_to_bf16_rne(f.x);
  o.y = f32_to_bf16_rne(f.y);
  o.z = f32_to_bf16_rne(f.z);
  o.w = f32_to_bf16_rne(f.w);
  ((ushort4*)dst)[i] = o;
}

// ---------------------------------------------------------------------------
// Generic WMMA GEMM: Y[M,Nout] = X[M,K] * W[Nout,K]^T + bias, bf16 inputs.
// Block = 256 threads = 8 waves, block tile 32(M) x 256(N); wave tile 16x64.
// MODE 0: gauge epilogue  -> xg bf16
// MODE 1: qkv scatter     -> q [B,H,N,D], k [B,H,N,D], vT [B,H,D,N] (bf16)
// MODE 2: proj            -> f32 out
// ---------------------------------------------------------------------------
template <int MODE>
__global__ void __launch_bounds__(256)
gemm_wmma(const unsigned short* __restrict__ Xbf,
          const unsigned short* __restrict__ Wbf,
          const float* __restrict__ bias,
          const float* __restrict__ xf32,   // MODE 0
          const float* __restrict__ theta,  // MODE 0
          unsigned short* __restrict__ out_a,   // MODE 0: xg, MODE 1: q
          unsigned short* __restrict__ out_k,   // MODE 1
          unsigned short* __restrict__ out_vt,  // MODE 1
          float* __restrict__ out_f32,          // MODE 2
          int M, int Nout, int K) {
  const int lane = threadIdx.x & 31;
  const int w    = threadIdx.x >> 5;
  const int lrow = lane & 15;   // A row / B column / D column
  const int lhi  = lane >> 4;   // half-wave selector
  const int m0 = blockIdx.x * 32 + (w >> 2) * 16;
  const int n0 = blockIdx.y * 256 + (w & 3) * 64;

  v8f acc[4] = {};

  const unsigned short* xrow = Xbf + (size_t)(m0 + lrow) * K;
  for (int k = 0; k < K; k += 32) {
    ABFrag a;
    a.h[0] = ld8(xrow + k + lhi * 8);        // K = kb .. kb+7
    a.h[1] = ld8(xrow + k + lhi * 8 + 16);   // K = kb+16 .. kb+23
    __builtin_prefetch(xrow + k + 64, 0, 0); // global_prefetch_b8
#pragma unroll
    for (int t = 0; t < 4; ++t) {
      const unsigned short* wr =
          Wbf + (size_t)(n0 + t * 16 + lrow) * K + k + lhi * 8;
      ABFrag b;
      b.h[0] = ld8(wr);       // K = kb .. kb+7
      b.h[1] = ld8(wr + 16);  // K = kb+16 .. kb+23
      acc[t] = wmma_bf16(a, b, acc[t]);
    }
  }

#pragma unroll
  for (int t = 0; t < 4; ++t) {
#pragma unroll
    for (int r = 0; r < 8; ++r) {
      const int m = m0 + r + 8 * lhi;        // D-layout row
      const int n = n0 + t * 16 + lrow;      // D-layout col
      float val = acc[t][r] + bias[n];
      if constexpr (MODE == 0) {
        float xv = xf32[(size_t)m * Nout + n];
        float xg = xv + 0.1f * val * theta[n];
        out_a[(size_t)m * Nout + n] = f32_to_bf16_rne(xg);
      } else if constexpr (MODE == 1) {
        const int which = n >> 10;          // 0:q 1:k 2:v
        const int rem = n & 1023;
        const int hh = rem >> 6;
        const int dd = rem & 63;
        const int bb = m >> 11;             // N = 2048
        const int nn = m & 2047;
        const unsigned short bv = f32_to_bf16_rne(val);
        if (which == 0)
          out_a[(((size_t)(bb * HH + hh) * NN + nn) * DD) + dd] = bv;
        else if (which == 1)
          out_k[(((size_t)(bb * HH + hh) * NN + nn) * DD) + dd] = bv;
        else
          out_vt[(((size_t)(bb * HH + hh) * DD + dd) * NN) + nn] = bv;
      } else {
        out_f32[(size_t)m * Nout + n] = val;
      }
    }
  }
}

// ---------------------------------------------------------------------------
// Flash-style attention: one block (4 waves / 128 threads) per (b, h, qTile16).
// Keys streamed in chunks of 512 with online softmax; S/P bounced via LDS to
// convert WMMA C/D layout -> A-fragment layout for the P*V matmul.
// q tile is staged into LDS via the CDNA5 async global->LDS path (ASYNCcnt).
// ---------------------------------------------------------------------------
__global__ void __launch_bounds__(128)
attn_wmma(const unsigned short* __restrict__ qb,
          const unsigned short* __restrict__ kb,
          const unsigned short* __restrict__ vtb,
          unsigned short* __restrict__ ob) {
  __shared__ __align__(16) float Ssc[16][512];            // 32 KB scores (chunk)
  __shared__ __align__(16) unsigned short Pbf[16][512];   // 16 KB probs (bf16)
  __shared__ __align__(16) unsigned short Qt[16][DD];     // 2 KB q tile
  __shared__ float s_mx[16], s_l[16], s_alpha[16];
  __shared__ float red[16][8];

  const int tid  = threadIdx.x;
  const int lane = tid & 31;
  const int w    = tid >> 5;
  const int lrow = lane & 15;
  const int lhi  = lane >> 4;

  const int q0 = blockIdx.x * 16;
  const int h  = blockIdx.y;
  const int b  = blockIdx.z;

  const unsigned short* qptr = qb  + ((size_t)(b * HH + h) * NN + q0) * DD;
  const unsigned short* kptr = kb  + ((size_t)(b * HH + h) * NN) * DD;
  const unsigned short* vptr = vtb + ((size_t)(b * HH + h) * DD) * NN;

  // --- async q-tile stage: 128 lanes x 16B = 16x64 bf16, tracked by ASYNCcnt
  {
    // LDS aperture: flat addr low 32 bits == LDS byte offset
    unsigned ldsoff = (unsigned)(size_t)(&Qt[0][0]) + (unsigned)tid * 16u;
    const unsigned short* gsrc = qptr + (size_t)tid * 8;
    asm volatile("global_load_async_to_lds_b128 %0, %1, off"
                 :: "v"(ldsoff), "v"(gsrc) : "memory");
    asm volatile("s_wait_asynccnt 0x0" ::: "memory");
  }
  if (tid < 16) { s_mx[tid] = -1e30f; s_l[tid] = 0.0f; }
  __syncthreads();

  // q A-fragments (constant across key chunks): kstep 0 (d 0..31), 1 (d 32..63)
  ABFrag a0, a1;
  a0.h[0] = *(const v8bf*)&Qt[lrow][lhi * 8];
  a0.h[1] = *(const v8bf*)&Qt[lrow][lhi * 8 + 16];
  a1.h[0] = *(const v8bf*)&Qt[lrow][32 + lhi * 8];
  a1.h[1] = *(const v8bf*)&Qt[lrow][32 + lhi * 8 + 16];

  v8f acco = {};

  for (int kc = 0; kc < 4; ++kc) {
    const int key0 = kc * 512;

    // ---- S = (q . k^T) * 0.125 for this wave's 128 keys ----
    for (int t = 0; t < 8; ++t) {
      const int jl = w * 128 + t * 16;
      const unsigned short* kr = kptr + (size_t)(key0 + jl + lrow) * DD + lhi * 8;
      ABFrag b0, b1;
      b0.h[0] = ld8(kr);      b0.h[1] = ld8(kr + 16);   // d 0..31 slice
      b1.h[0] = ld8(kr + 32); b1.h[1] = ld8(kr + 48);   // d 32..63 slice
      v8f s = {};
      s = wmma_bf16(a0, b0, s);
      s = wmma_bf16(a1, b1, s);
#pragma unroll
      for (int r = 0; r < 8; ++r)
        Ssc[r + 8 * lhi][jl + lrow] = s[r] * 0.125f;
    }
    __syncthreads();

    // ---- online softmax over the 512-key chunk ----
    const int row = tid & 15;
    const int seg = tid >> 4;       // 8 segments x 64 keys
    const int base = seg * 64;
    float lm = -1e30f;
    for (int i = 0; i < 64; ++i) lm = fmaxf(lm, Ssc[row][base + i]);
    red[row][seg] = lm;
    __syncthreads();
    if (tid < 16) {
      float cm = red[tid][0];
      for (int s2 = 1; s2 < 8; ++s2) cm = fmaxf(cm, red[tid][s2]);
      float nm = fmaxf(s_mx[tid], cm);
      s_alpha[tid] = __expf(s_mx[tid] - nm);
      s_mx[tid] = nm;
    }
    __syncthreads();
    const float rm = s_mx[row];
    float ls = 0.0f;
    for (int i = 0; i < 64; ++i) {
      float p = __expf(Ssc[row][base + i] - rm);
      Pbf[row][base + i] = f32_to_bf16_rne(p);
      ls += p;
    }
    red[row][seg] = ls;
    __syncthreads();
    if (tid < 16) {
      float al = 0.0f;
      for (int s2 = 0; s2 < 8; ++s2) al += red[tid][s2];
      s_l[tid] = s_l[tid] * s_alpha[tid] + al;
    }
    __syncthreads();

    // ---- rescale O, then O += P * V^T (this wave owns d-cols w*16..w*16+15)
#pragma unroll
    for (int r = 0; r < 8; ++r) acco[r] *= s_alpha[r + 8 * lhi];
    for (int js = 0; js < 16; ++js) {
      ABFrag pa;
      pa.h[0] = *(const v8bf*)&Pbf[lrow][js * 32 + lhi * 8];
      pa.h[1] = *(const v8bf*)&Pbf[lrow][js * 32 + lhi * 8 + 16];
      const unsigned short* vr =
          vptr + (size_t)(w * 16 + lrow) * NN + key0 + js * 32 + lhi * 8;
      ABFrag vb;
      vb.h[0] = ld8(vr); vb.h[1] = ld8(vr + 16);
      acco = wmma_bf16(pa, vb, acco);
    }
    __syncthreads();
  }

  // ---- finalize: O /= l, write [B,N,C] bf16 (col = h*64 + d) ----
#pragma unroll
  for (int r = 0; r < 8; ++r) {
    const int m = r + 8 * lhi;
    const float o = acco[r] / s_l[m];
    const int rowg = b * NN + q0 + m;
    const int colg = h * DD + w * 16 + lrow;
    ob[(size_t)rowg * CC + colg] = f32_to_bf16_rne(o);
  }
}

// ---------------------------------------------------------------------------
extern "C" void kernel_launch(void* const* d_in, const int* in_sizes, int n_in,
                              void* d_out, int out_size, void* d_ws, size_t ws_size,
                              hipStream_t stream) {
  const float* x     = (const float*)d_in[0];
  const float* theta = (const float*)d_in[1];
  const float* Wg    = (const float*)d_in[2];
  const float* bg    = (const float*)d_in[3];
  const float* Wqkv  = (const float*)d_in[4];
  const float* bqkv  = (const float*)d_in[5];
  const float* Wproj = (const float*)d_in[6];
  const float* bproj = (const float*)d_in[7];
  float* out = (float*)d_out;

  char* ws = (char*)d_ws;
  unsigned short* xbf     = (unsigned short*)(ws + 0);          //  8 MB
  unsigned short* wgbf    = (unsigned short*)(ws + 8388608);    //  2 MB
  unsigned short* wqkvbf  = (unsigned short*)(ws + 10485760);   //  6 MB
  unsigned short* wprojbf = (unsigned short*)(ws + 16777216);   //  2 MB
  unsigned short* xgbf    = (unsigned short*)(ws + 18874368);   //  8 MB
  unsigned short* qbuf    = (unsigned short*)(ws + 27262976);   //  8 MB
  unsigned short* kbuf    = (unsigned short*)(ws + 35651584);   //  8 MB
  unsigned short* vtbuf   = (unsigned short*)(ws + 44040192);   //  8 MB
  unsigned short* obuf    = (unsigned short*)(ws + 52428800);   //  8 MB

  // 1) f32 -> bf16 conversions
  {
    int n4;
    n4 = MM * CC / 4;
    cvt_f32_bf16<<<(n4 + 255) / 256, 256, 0, stream>>>(x, xbf, n4);
    n4 = CC * CC / 4;
    cvt_f32_bf16<<<(n4 + 255) / 256, 256, 0, stream>>>(Wg, wgbf, n4);
    n4 = 3 * CC * CC / 4;
    cvt_f32_bf16<<<(n4 + 255) / 256, 256, 0, stream>>>(Wqkv, wqkvbf, n4);
    n4 = CC * CC / 4;
    cvt_f32_bf16<<<(n4 + 255) / 256, 256, 0, stream>>>(Wproj, wprojbf, n4);
  }

  // 2) gauge GEMM: xg = x + 0.1 * ((x Wg^T + bg) * theta)
  gemm_wmma<0><<<dim3(MM / 32, CC / 256), 256, 0, stream>>>(
      xbf, wgbf, bg, x, theta, xgbf, nullptr, nullptr, nullptr, MM, CC, CC);

  // 3) qkv GEMM with scatter to q / k / v^T
  gemm_wmma<1><<<dim3(MM / 32, 3 * CC / 256), 256, 0, stream>>>(
      xgbf, wqkvbf, bqkv, nullptr, nullptr, qbuf, kbuf, vtbuf, nullptr,
      MM, 3 * CC, CC);

  // 4) attention: grid (N/16, H, B)
  attn_wmma<<<dim3(NN / 16, HH, BB), 128, 0, stream>>>(qbuf, kbuf, vtbuf, obuf);

  // 5) output projection -> f32 d_out
  gemm_wmma<2><<<dim3(MM / 32, CC / 256), 256, 0, stream>>>(
      obuf, wprojbf, bproj, nullptr, nullptr, nullptr, nullptr, nullptr, out,
      MM, CC, CC);
}
